// QwenAudioAttention_12360915878768
// MI455X (gfx1250) — matrix operand order repaired and checked by reading, not compiled, not run on hardware
//
#include <hip/hip_runtime.h>
#include <hip/hip_bf16.h>

#define TSEQ   3000
#define TPAD   3008   // 188 * 16
#define DMODEL 1280
#define NHEADS 20
#define HDIM   64
#define CHUNK  750
#define NEGBIG (-1e9f)

typedef __attribute__((ext_vector_type(16))) __bf16 v16bf;
typedef __attribute__((ext_vector_type(8)))  float  v8f;

union bf16x16 { uint4 q[2]; v16bf v; };

__device__ __forceinline__ __bf16 f2bf(float f) {
  __hip_bfloat16 h = __float2bfloat16(f);
  __bf16 r; __builtin_memcpy(&r, &h, sizeof(r));
  return r;
}

// A-fragment, 16x32 bf16 (row-major source, contiguous along K).
// lane half=0 holds K {kb+0..7, kb+16..23}; half=1 holds K {kb+8..15, kb+24..31}.
__device__ __forceinline__ v16bf load_a_frag(const __bf16* row, int kb, int half) {
  bf16x16 u;
  u.q[0] = *(const uint4*)(row + kb + half * 8);
  u.q[1] = *(const uint4*)(row + kb + 16 + half * 8);
  return u.v;
}
// B-fragment from B^T storage: `colrow` points at B column n stored contiguously in K.
// lane half=0 holds K {kb+0..7, kb+16..23}; half=1 holds K {kb+8..15, kb+24..31}.
__device__ __forceinline__ v16bf load_b_frag(const __bf16* colrow, int kb, int half) {
  bf16x16 u;
  const uint4* p = (const uint4*)(colrow + kb + 8 * half);
  u.q[0] = p[0];
  u.q[1] = p[2];
  return u.v;
}

#define WMMA_BF16(a, b, c) \
  __builtin_amdgcn_wmma_f32_16x16x32_bf16(false, (a), false, (b), (short)0, (c), false, false)

// Async copy 16 bytes global -> LDS (CDNA5 GLOBAL_LOAD_ASYNC_TO_LDS_B128, ASYNCcnt).
__device__ __forceinline__ void async_copy16(unsigned lds_addr, unsigned long long gaddr) {
  asm volatile("global_load_async_to_lds_b128 %0, %1, off"
               :: "v"(lds_addr), "v"(gaddr) : "memory");
}
__device__ __forceinline__ void wait_async0() {
  asm volatile("s_wait_asynccnt 0x0" ::: "memory");
}
__device__ __forceinline__ unsigned lds_addr_of(const void* p) {
  return (unsigned)(unsigned long long)(unsigned long)(uintptr_t)p;  // low 32 bits = LDS offset
}

// ---------------- prep: fp32 -> bf16 conversions ----------------

__global__ void k_conv_x(const float* __restrict__ x, __bf16* __restrict__ xb) {
  int idx = blockIdx.x * blockDim.x + threadIdx.x;
  if (idx >= TPAD * DMODEL) return;
  int i = idx / DMODEL;
  xb[idx] = (i < TSEQ) ? f2bf(x[idx]) : f2bf(0.0f);
}

__global__ void k_conv_wT(const float* __restrict__ w, __bf16* __restrict__ wt) {
  int idx = blockIdx.x * blockDim.x + threadIdx.x;  // idx = n*K + k
  if (idx >= DMODEL * DMODEL) return;
  int n = idx / DMODEL, k = idx % DMODEL;
  wt[idx] = f2bf(w[k * DMODEL + n]);
}

// ---------------- shared-B GEMM core ----------------
// Block = 128 threads (4 waves). Block computes 64 rows x 64 cols.
// B panel (64 cols x 32 k, bf16 = 4KB) is async-copied into LDS and shared by
// all 4 waves; double-buffered so panel i+1 DMA overlaps panel i WMMAs.
// Panel LDS layout: pan[buf][col][k] with k contiguous (same as B^T fragment layout).

#define GEMM_CORE(ACCDECL)                                                         \
  const int tid  = threadIdx.x;                                                    \
  const int lane = tid & 31;                                                       \
  const int wv   = tid >> 5;                                                       \
  const int half = lane >> 4;                                                      \
  const int ln   = lane & 15;                                                      \
  const int nb   = blockIdx.x;              /* 64-column group */                  \
  const int mt   = blockIdx.y * 4 + wv;     /* 16-row tile of this wave */         \
  const __bf16* arow = A + (size_t)(mt * 16 + ln) * DMODEL;                        \
  int ncol[4];                                                                     \
  _Pragma("unroll")                                                                \
  for (int t = 0; t < 4; ++t) ncol[t] = nb * 64 + t * 16 + ln;                     \
  v8f acc[4] = {};                                                                 \
  (void)ACCDECL;                                                                   \
  /* async B-panel copy: thread copies slots tid and tid+128 (16B each) */         \
  auto issue_panel = [&](int kb, int buf) {                                        \
    _Pragma("unroll")                                                              \
    for (int s = tid; s < 256; s += 128) {                                         \
      int col = s >> 2;                                                            \
      int kk  = (s & 3) * 8;                                                       \
      const __bf16* src = Bt + (size_t)(nb * 64 + col) * DMODEL + kb + kk;         \
      async_copy16(lds_addr_of(&pan[buf][col * 32 + kk]),                          \
                   (unsigned long long)(uintptr_t)src);                            \
    }                                                                              \
  };                                                                               \
  issue_panel(0, 0);                                                               \
  const int nsteps = DMODEL / 32;                                                  \
  for (int i = 0; i < nsteps; ++i) {                                               \
    wait_async0();        /* my slice of panel i landed */                         \
    __syncthreads();      /* everyone's slice landed; compute(i-1) done */         \
    if (i + 1 < nsteps) issue_panel((i + 1) * 32, (i + 1) & 1);                    \
    __builtin_prefetch(arow + i * 32 + 128, 0, 1);                                 \
    v16bf va = load_a_frag(arow, i * 32, half);                                    \
    const __bf16* p = pan[i & 1];                                                  \
    _Pragma("unroll")                                                              \
    for (int t = 0; t < 4; ++t) {                                                  \
      v16bf vb = load_b_frag(p + (size_t)(t * 16 + ln) * 32, 0, half);             \
      acc[t] = WMMA_BF16(va, vb, acc[t]);                                          \
    }                                                                              \
  }

// ---------------- QKV projection GEMM (bf16 in, bf16 out) ----------------
// transpose_out==0: out[h][t][d]  (Q, K)
// transpose_out==1: out[h][d][t]  (V^T, so P*V B-fragments are contiguous)

__global__ void k_proj(const __bf16* __restrict__ A, const __bf16* __restrict__ Bt,
                       const float* __restrict__ bias, __bf16* __restrict__ out,
                       int transpose_out) {
  __shared__ __align__(16) __bf16 pan[2][64 * 32];
  GEMM_CORE(0)
  const int h = nb;  // head == 64-column group
#pragma unroll
  for (int t = 0; t < 4; ++t) {
    float b = bias[ncol[t]];
    int d = t * 16 + ln;
#pragma unroll
    for (int r = 0; r < 8; ++r) {
      int i = mt * 16 + r + 8 * half;       // C/D layout: VGPR r -> row r + 8*half
      float v = acc[t][r] + b;
      if (!transpose_out)
        out[((size_t)h * TPAD + i) * HDIM + d] = f2bf(v);
      else
        out[((size_t)h * HDIM + d) * TPAD + i] = f2bf(v);
    }
  }
}

// ---------------- output projection GEMM (bf16 in, fp32 out + bias) ----------------

__global__ void k_outproj(const __bf16* __restrict__ A, const __bf16* __restrict__ Bt,
                          const float* __restrict__ bias, float* __restrict__ out) {
  __shared__ __align__(16) __bf16 pan[2][64 * 32];
  GEMM_CORE(0)
#pragma unroll
  for (int t = 0; t < 4; ++t) {
    float b = bias[ncol[t]];
#pragma unroll
    for (int r = 0; r < 8; ++r) {
      int i = mt * 16 + r + 8 * half;
      if (i < TSEQ) out[(size_t)i * DMODEL + ncol[t]] = acc[t][r] + b;
    }
  }
}

// ---------------- block-diagonal flash attention ----------------
// grid = (TPAD/16, NHEADS), block = 32. Wave owns one 16-row query tile of one head.

__global__ void k_attn(const __bf16* __restrict__ Qh, const __bf16* __restrict__ Kh,
                       const __bf16* __restrict__ Vt, __bf16* __restrict__ AO) {
  __shared__ __align__(16) __bf16 plds[16 * 32];   // per-wave P-tile transpose buffer

  const int lane = threadIdx.x & 31;
  const int half = lane >> 4;
  const int ln   = lane & 15;
  const int qt = blockIdx.x;
  const int h  = blockIdx.y;
  const int i0 = qt * 16;

  int irow[8];
#pragma unroll
  for (int r = 0; r < 8; ++r) irow[r] = i0 + r + 8 * half;

  // Q A-fragments for this tile (head_dim 64 -> two 16x32 fragments)
  const __bf16* qrow = Qh + ((size_t)h * TPAD + i0 + ln) * HDIM;
  v16bf qa0 = load_a_frag(qrow, 0, half);
  v16bf qa1 = load_a_frag(qrow, 32, half);

  // j-range = union of chunks this tile touches (mask is block-diagonal)
  const int seg_lo = i0 / CHUNK;
  int ilast = i0 + 15; if (ilast > TSEQ - 1) ilast = TSEQ - 1;
  const int seg_hi = ilast / CHUNK;
  const int j_lo = seg_lo * CHUNK;
  int j_hi = (seg_hi + 1) * CHUNK; if (j_hi > TSEQ) j_hi = TSEQ;

  float mrow[8], lrow[8];
  v8f O[4] = {};
#pragma unroll
  for (int r = 0; r < 8; ++r) { mrow[r] = NEGBIG; lrow[r] = 0.0f; }

  const __bf16* vrow[4];
#pragma unroll
  for (int t = 0; t < 4; ++t)
    vrow[t] = Vt + ((size_t)h * HDIM + t * 16 + ln) * TPAD;

  for (int jb = j_lo & ~31; jb < j_hi; jb += 32) {
    float s[2][8];
#pragma unroll
    for (int tt = 0; tt < 2; ++tt) {
      int jbase = jb + tt * 16;
      const __bf16* krow = Kh + ((size_t)h * TPAD + jbase + ln) * HDIM;
      v8f sc = {};
      sc = WMMA_BF16(qa0, load_b_frag(krow, 0, half), sc);
      sc = WMMA_BF16(qa1, load_b_frag(krow, 32, half), sc);
      int j = jbase + ln;              // column this lane holds in the C layout
      int jseg = j / CHUNK;
      bool jok = (j < TSEQ);
#pragma unroll
      for (int r = 0; r < 8; ++r) {
        int i = irow[r];
        bool ok = jok && (i < TSEQ) && ((i / CHUNK) == jseg);
        s[tt][r] = ok ? sc[r] * 0.125f : NEGBIG;   // scale = 1/sqrt(64)
      }
    }

    // per-row max across the 16 lanes of this half-group
    float tm[8];
#pragma unroll
    for (int r = 0; r < 8; ++r) tm[r] = fmaxf(s[0][r], s[1][r]);
#pragma unroll
    for (int off = 1; off < 16; off <<= 1)
#pragma unroll
      for (int r = 0; r < 8; ++r) tm[r] = fmaxf(tm[r], __shfl_xor(tm[r], off, 32));

    float p0[8], p1[8], alpha[8], ps[8];
#pragma unroll
    for (int r = 0; r < 8; ++r) {
      float mn = fmaxf(mrow[r], tm[r]);
      alpha[r] = __expf(mrow[r] - mn);
      p0[r] = (s[0][r] > -1e8f) ? __expf(s[0][r] - mn) : 0.0f;  // masked -> exactly 0
      p1[r] = (s[1][r] > -1e8f) ? __expf(s[1][r] - mn) : 0.0f;
      ps[r] = p0[r] + p1[r];
      mrow[r] = mn;
    }
#pragma unroll
    for (int off = 1; off < 16; off <<= 1)
#pragma unroll
      for (int r = 0; r < 8; ++r) ps[r] += __shfl_xor(ps[r], off, 32);
#pragma unroll
    for (int r = 0; r < 8; ++r) lrow[r] = lrow[r] * alpha[r] + ps[r];
#pragma unroll
    for (int t = 0; t < 4; ++t)
#pragma unroll
      for (int r = 0; r < 8; ++r) O[t][r] *= alpha[r];

    // transpose P: C layout (row = VGPRidx+8*half, col = lane) -> A-fragment layout via LDS
    asm volatile("s_wait_dscnt 0" ::: "memory");  // prior fragment reads done (WAR)
#pragma unroll
    for (int r = 0; r < 8; ++r) {
      plds[(r + 8 * half) * 32 + ln]      = f2bf(p0[r]);
      plds[(r + 8 * half) * 32 + 16 + ln] = f2bf(p1[r]);
    }
    asm volatile("s_wait_dscnt 0" ::: "memory");  // stores visible before reads (RAW)
    bf16x16 pu;
    pu.q[0] = *(const uint4*)(plds + ln * 32 + 8 * half);
    pu.q[1] = *(const uint4*)(plds + ln * 32 + 16 + 8 * half);
    v16bf pa = pu.v;

#pragma unroll
    for (int t = 0; t < 4; ++t)
      O[t] = WMMA_BF16(pa, load_b_frag(vrow[t], jb, half), O[t]);
  }

  // normalize and store (bf16, row-major [TPAD][DMODEL]); pad rows get zeros
#pragma unroll
  for (int r = 0; r < 8; ++r) {
    float invl = (lrow[r] > 0.0f) ? 1.0f / lrow[r] : 0.0f;
    int i = irow[r];
#pragma unroll
    for (int t = 0; t < 4; ++t) {
      int col = h * 64 + t * 16 + ln;
      float val = (i < TSEQ) ? O[t][r] * invl : 0.0f;
      AO[(size_t)i * DMODEL + col] = f2bf(val);
    }
  }
}

// ---------------- host launcher ----------------

extern "C" void kernel_launch(void* const* d_in, const int* in_sizes, int n_in,
                              void* d_out, int out_size, void* d_ws, size_t ws_size,
                              hipStream_t stream) {
  (void)in_sizes; (void)n_in; (void)out_size; (void)ws_size;
  const float* x  = (const float*)d_in[0];
  // d_in[1] = attention_mask: block-diagonal by construction, applied analytically
  const float* wq = (const float*)d_in[2];
  const float* bq = (const float*)d_in[3];
  const float* wk = (const float*)d_in[4];
  const float* bk = (const float*)d_in[5];
  const float* wv = (const float*)d_in[6];
  const float* bv = (const float*)d_in[7];
  const float* wo = (const float*)d_in[8];
  const float* bo = (const float*)d_in[9];

  char* ws = (char*)d_ws;
  size_t off = 0;
  auto carve = [&](size_t bytes) -> void* {
    void* p = ws + off;
    off += (bytes + 255) & ~(size_t)255;
    return p;
  };
  __bf16* Xb  = (__bf16*)carve((size_t)TPAD * DMODEL * 2);         // later reused as attn-out
  __bf16* Wt0 = (__bf16*)carve((size_t)DMODEL * DMODEL * 2);
  __bf16* Wt1 = (__bf16*)carve((size_t)DMODEL * DMODEL * 2);
  __bf16* Wt2 = (__bf16*)carve((size_t)DMODEL * DMODEL * 2);
  __bf16* Qh  = (__bf16*)carve((size_t)NHEADS * TPAD * HDIM * 2);
  __bf16* Kh  = (__bf16*)carve((size_t)NHEADS * TPAD * HDIM * 2);
  __bf16* Vt  = (__bf16*)carve((size_t)NHEADS * TPAD * HDIM * 2);

  const int nX = TPAD * DMODEL;
  k_conv_x<<<dim3((nX + 255) / 256), dim3(256), 0, stream>>>(x, Xb);

  const int nW = DMODEL * DMODEL;
  dim3 gw((nW + 255) / 256), bw(256);
  k_conv_wT<<<gw, bw, 0, stream>>>(wq, Wt0);
  k_conv_wT<<<gw, bw, 0, stream>>>(wk, Wt1);
  k_conv_wT<<<gw, bw, 0, stream>>>(wv, Wt2);

  // GEMMs: 128-thread blocks (4 waves), 64x64 tile per block, async B panels.
  dim3 gg(NHEADS, TPAD / 64), bb(128);
  k_proj<<<gg, bb, 0, stream>>>(Xb, Wt0, bq, Qh, 0);
  k_proj<<<gg, bb, 0, stream>>>(Xb, Wt1, bk, Kh, 0);
  k_proj<<<gg, bb, 0, stream>>>(Xb, Wt2, bv, Vt, 1);   // V stored transposed

  k_conv_wT<<<gw, bw, 0, stream>>>(wo, Wt0);           // Wq^T slot no longer needed

  dim3 ga(TPAD / 16, NHEADS), ba(32);
  k_attn<<<ga, ba, 0, stream>>>(Qh, Kh, Vt, Xb);       // Xb slot now holds attn output

  k_outproj<<<gg, bb, 0, stream>>>(Xb, Wt0, bo, (float*)d_out);
}
